// Net_70643622084791
// MI455X (gfx1250) — matrix operand order repaired and checked by reading, not compiled, not run on hardware
//
#include <hip/hip_runtime.h>

// ---------------------------------------------------------------------------
// CDNA5 (gfx1250) weighted-GIN network.
// GEMMs use v_wmma_f32_16x16x32_f16 (f16 operands, f32 accumulate).
// One wave computes a 16x64 output row-block: A frags reused across 4
// independent WMMA accumulator chains; B read as contiguous 128B rows.
// Scatter/pool/BN-stats use hardware global_atomic_add_f32 (L2-resident).
// ---------------------------------------------------------------------------

typedef __attribute__((ext_vector_type(16))) _Float16 v16h;
typedef __attribute__((ext_vector_type(8)))  float    v8f;
typedef __attribute__((ext_vector_type(4)))  float    v4f;

union HFrag { v4f f[2]; v16h h; };

// ---------------- utility kernels ----------------

__global__ __launch_bounds__(256) void zero_kernel(float* __restrict__ p, long long n4) {
    long long i = (long long)blockIdx.x * blockDim.x + threadIdx.x;
    if (i < n4) {
        v4f z = {0.f, 0.f, 0.f, 0.f};
        reinterpret_cast<v4f*>(p)[i] = z;
    }
}

__global__ __launch_bounds__(256) void cvt_f16_kernel(const float* __restrict__ in,
                                                      _Float16* __restrict__ out, int n) {
    int i = blockIdx.x * blockDim.x + threadIdx.x;
    if (i < n) out[i] = (_Float16)in[i];
}

// z = f16(a + b), elementwise
__global__ __launch_bounds__(256) void add_cvt_kernel(const float* __restrict__ a,
                                                      const float* __restrict__ b,
                                                      _Float16* __restrict__ out, long long n) {
    long long i = (long long)blockIdx.x * blockDim.x + threadIdx.x;
    if (i < n) out[i] = (_Float16)(a[i] + b[i]);
}

// ---------------- edge gather/scale/scatter (weighted GIN aggregation) ------
// agg[dst] += ew * h[src]; FVEC = F/4 float4 chunks per edge.
template <int FVEC>
__global__ __launch_bounds__(256) void edge_agg_kernel(const int* __restrict__ src,
                                                       const int* __restrict__ dst,
                                                       const float* __restrict__ ew,
                                                       const float* __restrict__ h,
                                                       float* __restrict__ agg, int E) {
    constexpr int EPB = 256 / FVEC;
    int e  = blockIdx.x * EPB + (int)(threadIdx.x / FVEC);
    int f4 = (int)(threadIdx.x % FVEC);
    if (e >= E) return;
    int s = src[e], d = dst[e];
    float w = ew[e];
    v4f v = reinterpret_cast<const v4f*>(h + (size_t)s * (FVEC * 4))[f4];
    float* out = agg + (size_t)d * (FVEC * 4) + f4 * 4;
    atomicAdd(out + 0, w * v.x);
    atomicAdd(out + 1, w * v.y);
    atomicAdd(out + 2, w * v.z);
    atomicAdd(out + 3, w * v.w);
}

// ---------------- WMMA GEMM: out = relu(A[M,K] @ B[K,64] + bias) ------------
// One wave per 16-row block; computes all 4 column tiles of the 64-wide
// output. A fragments are loaded once and reused by 4 independent WMMA
// accumulator chains (better ILP, 4x less A traffic). Ncols fixed at 64.
template <int K, bool OUTF, bool OUTH>
__global__ __launch_bounds__(256) void wmma_gemm_kernel(const _Float16* __restrict__ A,
                                                        const _Float16* __restrict__ B,
                                                        const float* __restrict__ bias,
                                                        float* __restrict__ outF,
                                                        _Float16* __restrict__ outH,
                                                        int M) {
    int wave = (int)((blockIdx.x * blockDim.x + threadIdx.x) >> 5);
    int lane = (int)(threadIdx.x & 31);
    int mt = wave;
    if (mt * 16 >= M) return;                 // wave-uniform: EXEC stays all-1s

    int m  = lane & 15;                       // A row within tile
    int hi = lane >> 4;                       // A K-half select
    const _Float16* Arow = A + (size_t)(mt * 16 + m) * K + hi * 8;

    v8f c[4] = {{}, {}, {}, {}};
#pragma unroll
    for (int k0 = 0; k0 < K; k0 += 32) {
        // A 16x32 frag: slots 0-7 = K k0+hi*8.., slots 8-15 = K k0+16+hi*8..
        HFrag a;
        const v4f* pa = reinterpret_cast<const v4f*>(Arow + k0);
        a.f[0] = pa[0];
        a.f[1] = pa[2];
        // B frag: column n = lane&15; slots 0-7 = K k0+hi*8.., slots 8-15 = K k0+16+hi*8..
        const _Float16* pb = B + (size_t)(k0 + hi * 8) * 64 + (lane & 15);
        HFrag b[4];
#pragma unroll
        for (int t = 0; t < 4; ++t) {
#pragma unroll
            for (int s = 0; s < 8; ++s) {
                b[t].h[s]     = pb[(size_t)s * 64 + t * 16];
                b[t].h[8 + s] = pb[(size_t)(16 + s) * 64 + t * 16];
            }
        }
#pragma unroll
        for (int t = 0; t < 4; ++t) {
            c[t] = __builtin_amdgcn_wmma_f32_16x16x32_f16(false, a.h, false, b[t].h,
                                                          (short)0, c[t], false, false);
        }
    }

    // D layout: n = lane&15 ; m = (lane>>4)*8 + v
    int n     = lane & 15;
    int mbase = mt * 16 + hi * 8;
#pragma unroll
    for (int t = 0; t < 4; ++t) {
        float bv = bias[t * 16 + n];
#pragma unroll
        for (int v = 0; v < 8; ++v) {
            float val = fmaxf(c[t][v] + bv, 0.0f);
            size_t o = (size_t)(mbase + v) * 64 + t * 16 + n;
            if (OUTF) outF[o] = val;
            if (OUTH) outH[o] = (_Float16)val;
        }
    }
}

// ---------------- BatchNorm statistics (64 columns) -------------------------
__global__ __launch_bounds__(256) void bn_stats_kernel(const float* __restrict__ r,
                                                       int Nrows, float* __restrict__ stats) {
    __shared__ float ssum[64], ssq[64];
    int t = threadIdx.x;
    if (t < 64) { ssum[t] = 0.f; ssq[t] = 0.f; }
    __syncthreads();
    int c      = t & 63;
    int walker = (int)((blockIdx.x * 256 + t) >> 6);
    int nwalk  = (int)((gridDim.x * 256) >> 6);
    float s = 0.f, q = 0.f;
    for (int row = walker; row < Nrows; row += nwalk) {
        float v = r[(size_t)row * 64 + c];
        s += v; q += v * v;
    }
    atomicAdd(&ssum[c], s);
    atomicAdd(&ssq[c], q);
    __syncthreads();
    if (t < 64) {
        atomicAdd(&stats[t],      ssum[t]);
        atomicAdd(&stats[64 + t], ssq[t]);
    }
}

__global__ void bn_final_kernel(const float* __restrict__ stats,
                                const float* __restrict__ gamma,
                                const float* __restrict__ beta,
                                float* __restrict__ scsh, float invN) {
    int c = threadIdx.x;  // 64
    float mean = stats[c] * invN;
    float var  = stats[64 + c] * invN - mean * mean;
    float sc   = gamma[c] * rsqrtf(var + 1e-5f);
    scsh[c]      = sc;
    scsh[64 + c] = beta[c] - mean * sc;
}

// h = r*scale + shift ; pool[batch[row]] += h ; optionally store h.
template <bool STORE_H>
__global__ __launch_bounds__(256) void bn_apply_pool_kernel(const float* __restrict__ r,
                                                            const float* __restrict__ scsh,
                                                            const int* __restrict__ batch,
                                                            float* __restrict__ hout,
                                                            float* __restrict__ pool,
                                                            int Nrows) {
    long long idx = (long long)blockIdx.x * blockDim.x + threadIdx.x;
    if (idx >= (long long)Nrows * 64) return;
    int row = (int)(idx >> 6);
    int c   = (int)(idx & 63);
    float v = r[idx] * scsh[c] + scsh[64 + c];
    if (STORE_H) hout[idx] = v;
    atomicAdd(&pool[(size_t)batch[row] * 64 + c], v);
}

// g_f16[G,128] = concat(p0, p1)
__global__ __launch_bounds__(256) void concat_kernel(const float* __restrict__ p0,
                                                     const float* __restrict__ p1,
                                                     _Float16* __restrict__ g, int G) {
    int i = blockIdx.x * blockDim.x + threadIdx.x;
    if (i >= G * 128) return;
    int row = i >> 7;
    int c   = i & 127;
    float v = (c < 64) ? p0[row * 64 + c] : p1[row * 64 + (c - 64)];
    g[i] = (_Float16)v;
}

// logits = t2[G,64] @ wf2[64,2] + bf2 ; out = log_softmax
__global__ __launch_bounds__(256) void head_final_kernel(const float* __restrict__ t2,
                                                         const float* __restrict__ wf2,
                                                         const float* __restrict__ bf2,
                                                         float* __restrict__ out, int G) {
    int g = blockIdx.x * blockDim.x + threadIdx.x;
    if (g >= G) return;
    float l0 = bf2[0], l1 = bf2[1];
#pragma unroll 8
    for (int k = 0; k < 64; ++k) {
        float v = t2[(size_t)g * 64 + k];
        l0 += v * wf2[k * 2 + 0];
        l1 += v * wf2[k * 2 + 1];
    }
    float m   = fmaxf(l0, l1);
    float lse = m + __logf(__expf(l0 - m) + __expf(l1 - m));
    out[g * 2 + 0] = l0 - lse;
    out[g * 2 + 1] = l1 - lse;
}

// ---------------------------------------------------------------------------

extern "C" void kernel_launch(void* const* d_in, const int* in_sizes, int n_in,
                              void* d_out, int out_size, void* d_ws, size_t ws_size,
                              hipStream_t stream) {
    (void)n_in; (void)ws_size;

    const float* x     = (const float*)d_in[0];
    const int*   ei    = (const int*)d_in[1];
    const float* ew    = (const float*)d_in[2];
    const int*   batch = (const int*)d_in[3];
    const float* w1a = (const float*)d_in[4];  const float* b1a = (const float*)d_in[5];
    const float* w2a = (const float*)d_in[6];  const float* b2a = (const float*)d_in[7];
    const float* g0  = (const float*)d_in[8];  const float* be0 = (const float*)d_in[9];
    const float* w1b = (const float*)d_in[10]; const float* b1b = (const float*)d_in[11];
    const float* w2b = (const float*)d_in[12]; const float* b2b = (const float*)d_in[13];
    const float* g1  = (const float*)d_in[14]; const float* be1 = (const float*)d_in[15];
    const float* wf  = (const float*)d_in[16]; const float* bf  = (const float*)d_in[17];
    const float* wf1 = (const float*)d_in[18]; const float* bf1 = (const float*)d_in[19];
    const float* wf2 = (const float*)d_in[20]; const float* bf2 = (const float*)d_in[21];
    float* out = (float*)d_out;

    const int N = in_sizes[3];            // 100000 nodes
    const int E = in_sizes[2];            // 1600000 edges
    const int F = in_sizes[0] / N;        // 128
    const int D = in_sizes[5];            // 64
    const int G = out_size / 2;           // 512 graphs
    const int* src = ei;
    const int* dst = ei + E;

    // ---- workspace carve-out (256B aligned) ----
    char* base = (char*)d_ws;
    size_t off = 0;
    auto take = [&](size_t bytes) -> char* {
        char* p = base + off;
        off += (bytes + 255) & ~(size_t)255;
        return p;
    };
    float*    agg0  = (float*)take((size_t)N * F * 4);
    _Float16* z0h   = (_Float16*)take((size_t)N * F * 2);
    _Float16* t0h   = (_Float16*)take((size_t)N * D * 2);
    float*    r0    = (float*)take((size_t)N * D * 4);
    float*    h0    = (float*)take((size_t)N * D * 4);
    float*    agg1  = (float*)take((size_t)N * D * 4);
    _Float16* z1h   = (_Float16*)take((size_t)N * D * 2);
    _Float16* t1h   = (_Float16*)take((size_t)N * D * 2);
    float*    r1    = (float*)take((size_t)N * D * 4);
    float*    stats = (float*)take(512 * 4);     // [stats0|scsh0|stats1|scsh1]
    float*    p0    = (float*)take((size_t)G * D * 4);  // contiguous after stats
    float*    p1    = (float*)take((size_t)G * D * 4);
    _Float16* gh    = (_Float16*)take((size_t)G * 2 * D * 2);
    _Float16* th1   = (_Float16*)take((size_t)G * D * 2);
    float*    t2    = (float*)take((size_t)G * D * 4);
    _Float16* w1ah  = (_Float16*)take((size_t)F * D * 2);
    _Float16* w2ah  = (_Float16*)take((size_t)D * D * 2);
    _Float16* w1bh  = (_Float16*)take((size_t)D * D * 2);
    _Float16* w2bh  = (_Float16*)take((size_t)D * D * 2);
    _Float16* wfh   = (_Float16*)take((size_t)2 * D * D * 2);
    _Float16* wf1h  = (_Float16*)take((size_t)D * D * 2);
    float* stats0 = stats;
    float* scsh0  = stats + 128;
    float* stats1 = stats + 256;
    float* scsh1  = stats + 384;

    auto blks = [](long long n, int b) { return (unsigned)((n + b - 1) / b); };

    // ---- weight f32 -> f16 conversion ----
    cvt_f16_kernel<<<blks(F * D, 256), 256, 0, stream>>>(w1a, w1ah, F * D);
    cvt_f16_kernel<<<blks(D * D, 256), 256, 0, stream>>>(w2a, w2ah, D * D);
    cvt_f16_kernel<<<blks(D * D, 256), 256, 0, stream>>>(w1b, w1bh, D * D);
    cvt_f16_kernel<<<blks(D * D, 256), 256, 0, stream>>>(w2b, w2bh, D * D);
    cvt_f16_kernel<<<blks(2 * D * D, 256), 256, 0, stream>>>(wf, wfh, 2 * D * D);
    cvt_f16_kernel<<<blks(D * D, 256), 256, 0, stream>>>(wf1, wf1h, D * D);

    // ---- zero accumulators (every call: deterministic under graph replay) ----
    zero_kernel<<<blks((long long)N * F / 4, 256), 256, 0, stream>>>(agg0, (long long)N * F / 4);
    zero_kernel<<<blks((long long)N * D / 4, 256), 256, 0, stream>>>(agg1, (long long)N * D / 4);
    // stats + p0 + p1 are contiguous: 512 + 2*G*D floats
    zero_kernel<<<blks((512 + 2LL * G * D) / 4, 256), 256, 0, stream>>>(
        stats, (512 + 2LL * G * D) / 4);

    // one wave per 16-row block (all 64 output columns per wave)
    const unsigned GEMM_BLOCKS_N = blks(N / 16, 8);
    const unsigned GEMM_BLOCKS_G = blks(G / 16, 8);

    // ================= layer 0 =================
    edge_agg_kernel<32><<<blks(E, 8), 256, 0, stream>>>(src, dst, ew, x, agg0, E);
    add_cvt_kernel<<<blks((long long)N * F, 256), 256, 0, stream>>>(x, agg0, z0h, (long long)N * F);
    wmma_gemm_kernel<128, false, true><<<GEMM_BLOCKS_N, 256, 0, stream>>>(z0h, w1ah, b1a, nullptr, t0h, N);
    wmma_gemm_kernel<64, true, false><<<GEMM_BLOCKS_N, 256, 0, stream>>>(t0h, w2ah, b2a, r0, nullptr, N);
    bn_stats_kernel<<<256, 256, 0, stream>>>(r0, N, stats0);
    bn_final_kernel<<<1, 64, 0, stream>>>(stats0, g0, be0, scsh0, 1.0f / (float)N);
    bn_apply_pool_kernel<true><<<blks((long long)N * D, 256), 256, 0, stream>>>(r0, scsh0, batch, h0, p0, N);

    // ================= layer 1 =================
    edge_agg_kernel<16><<<blks(E, 16), 256, 0, stream>>>(src, dst, ew, h0, agg1, E);
    add_cvt_kernel<<<blks((long long)N * D, 256), 256, 0, stream>>>(h0, agg1, z1h, (long long)N * D);
    wmma_gemm_kernel<64, false, true><<<GEMM_BLOCKS_N, 256, 0, stream>>>(z1h, w1bh, b1b, nullptr, t1h, N);
    wmma_gemm_kernel<64, true, false><<<GEMM_BLOCKS_N, 256, 0, stream>>>(t1h, w2bh, b2b, r1, nullptr, N);
    bn_stats_kernel<<<256, 256, 0, stream>>>(r1, N, stats1);
    bn_final_kernel<<<1, 64, 0, stream>>>(stats1, g1, be1, scsh1, 1.0f / (float)N);
    bn_apply_pool_kernel<false><<<blks((long long)N * D, 256), 256, 0, stream>>>(r1, scsh1, batch, nullptr, p1, N);

    // ================= head =================
    concat_kernel<<<blks(G * 128, 256), 256, 0, stream>>>(p0, p1, gh, G);
    wmma_gemm_kernel<128, false, true><<<GEMM_BLOCKS_G, 256, 0, stream>>>(gh, wfh, bf, nullptr, th1, G);
    wmma_gemm_kernel<64, true, false><<<GEMM_BLOCKS_G, 256, 0, stream>>>(th1, wf1h, bf1, t2, nullptr, G);
    head_final_kernel<<<blks(G, 256), 256, 0, stream>>>(t2, wf2, bf2, out, G);
}
